// ESA_9380208575118
// MI455X (gfx1250) — matrix operand order repaired and checked by reading, compiled, not run on hardware
//
#include <hip/hip_runtime.h>

#define Bb   16
#define Mm   512
#define Dd   256
#define Hh   8
#define DHd  32
#define HIDh 512
#define NROWS (Bb * Mm)   // 8192

typedef __attribute__((ext_vector_type(16))) _Float16 v16h;
typedef __attribute__((ext_vector_type(8)))  _Float16 v8h;
typedef __attribute__((ext_vector_type(8)))  float    v8f;

static __device__ inline v8f wmma_f16f32(v16h a, v16h b, v8f c) {
  // 8 args: (neg_a, A, neg_b, B, c_mod, C, reuse_a, reuse_b)
  return __builtin_amdgcn_wmma_f32_16x16x32_f16(false, a, false, b, (short)0, c,
                                                false, false);
}

static __device__ inline v16h cat8(v8h lo, v8h hi) {
  return __builtin_shufflevector(lo, hi, 0, 1, 2, 3, 4, 5, 6, 7,
                                 8, 9, 10, 11, 12, 13, 14, 15);
}

static __device__ inline float gelu_tanh(float x) {
  float t = tanhf(0.7978845608028654f * (x + 0.044715f * x * x * x));
  return 0.5f * x * (1.f + t);
}

// ---------------- weight convert + transpose: f32 [K][N] -> f16 [N][K] --------
__global__ __launch_bounds__(256) void cvt_transpose(const float* __restrict__ s,
                                                     _Float16* __restrict__ d,
                                                     int K, int N) {
  int idx = blockIdx.x * 256 + threadIdx.x;
  if (idx < K * N) {
    int k = idx / N, n = idx - k * N;
    d[(size_t)n * K + k] = (_Float16)s[idx];
  }
}

// ---------------- LayerNorm over D=256, one wave per row ----------------
__global__ __launch_bounds__(256) void ln_kernel(const float* __restrict__ X,
                                                 const float* __restrict__ g,
                                                 const float* __restrict__ bta,
                                                 _Float16* __restrict__ out) {
  int wave = threadIdx.x >> 5, lane = threadIdx.x & 31;
  int row = blockIdx.x * 8 + wave;
  const float* x = X + (size_t)row * Dd;
  float v[8];
  float sum = 0.f;
#pragma unroll
  for (int i = 0; i < 8; ++i) { v[i] = x[lane + i * 32]; sum += v[i]; }
#pragma unroll
  for (int m = 1; m < 32; m <<= 1) sum += __shfl_xor(sum, m, 32);
  float mean = sum * (1.f / Dd);
  float var = 0.f;
#pragma unroll
  for (int i = 0; i < 8; ++i) { float dd = v[i] - mean; var += dd * dd; }
#pragma unroll
  for (int m = 1; m < 32; m <<= 1) var += __shfl_xor(var, m, 32);
  float inv = rsqrtf(var * (1.f / Dd) + 1e-5f);
#pragma unroll
  for (int i = 0; i < 8; ++i) {
    int c = lane + i * 32;
    out[(size_t)row * Dd + c] = (_Float16)((v[i] - mean) * inv * g[c] + bta[c]);
  }
}

// ---------------- WMMA GEMM: one 16x64 strip per wave (4 N-tiles) -------------
// A: [8192 x K] f16 row-major. Wt: weights TRANSPOSED [N][K] f16 row-major,
// so B fragments are two contiguous b128 loads.
// EPI: 0 = bias -> f16 out ; 1 = bias+GELU -> f16 out ; 2 = bias+resid -> f32 out
template <int EPI, int K>
__global__ __launch_bounds__(256) void gemm16(const _Float16* __restrict__ A,
                                              const _Float16* __restrict__ Wt,
                                              const float* __restrict__ bias,
                                              const float* __restrict__ resid,
                                              void* __restrict__ Cout, int N) {
  int ncg = N >> 6;  // 64-wide column groups
  int t = blockIdx.x * 8 + (threadIdx.x >> 5);
  int row0 = (t / ncg) << 4;
  int colg = (t % ncg) << 6;
  int lane = threadIdx.x & 31;
  int hi = lane >> 4, ln = lane & 15;
  v8f acc[4];
#pragma unroll
  for (int c = 0; c < 4; ++c)
#pragma unroll
    for (int i = 0; i < 8; ++i) acc[c][i] = 0.f;
#pragma unroll 4
  for (int k0 = 0; k0 < K; k0 += 32) {
    // A fragment: row = ln ; K = hi*16 + 0..15 (contiguous)
    const _Float16* ap = A + (size_t)(row0 + ln) * K + k0 + hi * 16;
    v16h av = cat8(*(const v8h*)ap, *(const v8h*)(ap + 8));
#pragma unroll
    for (int c = 0; c < 4; ++c) {
      // B fragment: col = colg + c*16 + ln ; K = k0 + hi*16 + e  (contiguous)
      const _Float16* bp = Wt + (size_t)(colg + c * 16 + ln) * K + k0 + hi * 16;
      v16h bv = cat8(*(const v8h*)bp, *(const v8h*)(bp + 8));
      acc[c] = wmma_f16f32(av, bv, acc[c]);
    }
  }
#pragma unroll
  for (int c = 0; c < 4; ++c) {
    int col = colg + c * 16 + ln;
    float bc = bias[col];
#pragma unroll
    for (int r = 0; r < 8; ++r) {
      int row = row0 + hi * 8 + r;
      float v = acc[c][r] + bc;
      if (EPI == 1) v = gelu_tanh(v);
      size_t off = (size_t)row * N + col;
      if (EPI == 2)
        ((float*)Cout)[off] = v + resid[off];
      else
        ((_Float16*)Cout)[off] = (_Float16)v;
    }
  }
}

// ---------------- fused masked attention, flash-style ----------------
// grid = B*H*8 blocks of 128 threads; each wave owns 16 query rows.
__global__ __launch_bounds__(128) void attn_kernel(const _Float16* __restrict__ Qg,
                                                   const _Float16* __restrict__ Kg,
                                                   const _Float16* __restrict__ Vg,
                                                   const int* __restrict__ srcI,
                                                   const int* __restrict__ tgtI,
                                                   _Float16* __restrict__ Og) {
  __shared__ _Float16 Vt[DHd][Mm];       // V transposed: [dh][key] (32 KB)
  __shared__ _Float16 Pbuf[4][16][32];   // per-wave P staging (4 KB)

  int blk = blockIdx.x;
  int qb = blk & 7;
  int h = (blk >> 3) & (Hh - 1);
  int b = blk >> 6;
  int ebase = b * Mm;
  int tid = threadIdx.x;
  int wave = tid >> 5, lane = tid & 31;
  int hi = lane >> 4, ln = lane & 15;

  for (int idx = tid; idx < Mm * DHd; idx += 128) {
    int key = idx >> 5, dh = idx & 31;
    Vt[dh][key] = Vg[(size_t)(ebase + key) * Dd + h * DHd + dh];
  }
  __syncthreads();

  int qrow0 = qb * 64 + wave * 16;
  v16h qa;
  {
    const _Float16* qp = Qg + (size_t)(ebase + qrow0 + ln) * Dd + h * DHd + hi * 16;
    qa = cat8(*(const v8h*)qp, *(const v8h*)(qp + 8));
  }
  int sq[8], tq[8];
#pragma unroll
  for (int r = 0; r < 8; ++r) {
    int e = ebase + qrow0 + hi * 8 + r;
    sq[r] = srcI[e];
    tq[r] = tgtI[e];
  }
  float mrow[8], ssum[8];
  v8f O0, O1;
#pragma unroll
  for (int r = 0; r < 8; ++r) { mrow[r] = -1e30f; ssum[r] = 0.f; O0[r] = 0.f; O1[r] = 0.f; }
  const float scale = 0.17677669529663687f;  // 1/sqrt(32)

  for (int p = 0; p < 16; ++p) {
    int k0 = p * 32;
    v16h kb0, kb1;
    {
      // K^T B-fragments: col = key (k0+ln / +16), K-dim = dh = hi*16 + e
      const _Float16* kp = Kg + (size_t)(ebase + k0 + ln) * Dd + h * DHd + hi * 16;
      kb0 = cat8(*(const v8h*)kp, *(const v8h*)(kp + 8));
      const _Float16* kp1 = kp + (size_t)16 * Dd;
      kb1 = cat8(*(const v8h*)kp1, *(const v8h*)(kp1 + 8));
    }
    v8f zf;
#pragma unroll
    for (int i = 0; i < 8; ++i) zf[i] = 0.f;
    v8f S0 = wmma_f16f32(qa, kb0, zf);
    v8f S1 = wmma_f16f32(qa, kb1, zf);

    int j0 = k0 + ln, j1 = j0 + 16;
    int sk0 = srcI[ebase + j0], tk0 = tgtI[ebase + j0];
    int sk1 = srcI[ebase + j1], tk1 = tgtI[ebase + j1];
    float s0v[8], s1v[8];
#pragma unroll
    for (int r = 0; r < 8; ++r) {
      int irow = qrow0 + hi * 8 + r;
      // bitwise ops -> branchless v_cmp/v_cndmask, no EXEC gymnastics
      int a0 = ((int)(sq[r] == sk0) | (int)(tq[r] == tk0) | (int)(sq[r] == tk0) |
                (int)(tq[r] == sk0)) &
               (int)(irow != j0);
      int a1 = ((int)(sq[r] == sk1) | (int)(tq[r] == tk1) | (int)(sq[r] == tk1) |
                (int)(tq[r] == sk1)) &
               (int)(irow != j1);
      s0v[r] = S0[r] * scale + (a0 ? 0.f : -99999.f);
      s1v[r] = S1[r] * scale + (a1 ? 0.f : -99999.f);
    }
#pragma unroll
    for (int r = 0; r < 8; ++r) {
      float rm = fmaxf(s0v[r], s1v[r]);
      rm = fmaxf(rm, __shfl_xor(rm, 1, 16));
      rm = fmaxf(rm, __shfl_xor(rm, 2, 16));
      rm = fmaxf(rm, __shfl_xor(rm, 4, 16));
      rm = fmaxf(rm, __shfl_xor(rm, 8, 16));
      float mnew = fmaxf(mrow[r], rm);
      float f = __expf(mrow[r] - mnew);
      mrow[r] = mnew;
      float p0 = __expf(s0v[r] - mnew);
      float p1 = __expf(s1v[r] - mnew);
      ssum[r] = ssum[r] * f + p0 + p1;
      O0[r] *= f;
      O1[r] *= f;
      Pbuf[wave][hi * 8 + r][ln] = (_Float16)p0;
      Pbuf[wave][hi * 8 + r][16 + ln] = (_Float16)p1;
    }
    asm volatile("" ::: "memory");  // keep DS store->load order at IR level
    v16h pa, vb0, vb1;
    {
      // P A-fragment: row = ln, K = hi*16 + 0..15 (contiguous)
      pa = cat8(*(const v8h*)&Pbuf[wave][ln][hi * 16],
                *(const v8h*)&Pbuf[wave][ln][hi * 16 + 8]);
      // V B-fragments: col = dh (ln / 16+ln), K = key = k0 + hi*16 + e
      vb0 = cat8(*(const v8h*)&Vt[ln][k0 + hi * 16],
                 *(const v8h*)&Vt[ln][k0 + hi * 16 + 8]);
      vb1 = cat8(*(const v8h*)&Vt[16 + ln][k0 + hi * 16],
                 *(const v8h*)&Vt[16 + ln][k0 + hi * 16 + 8]);
    }
    O0 = wmma_f16f32(pa, vb0, O0);
    O1 = wmma_f16f32(pa, vb1, O1);
  }
#pragma unroll
  for (int r = 0; r < 8; ++r) {
    float s = ssum[r];
    s += __shfl_xor(s, 1, 16);
    s += __shfl_xor(s, 2, 16);
    s += __shfl_xor(s, 4, 16);
    s += __shfl_xor(s, 8, 16);
    float inv = 1.f / s;
    size_t off = (size_t)(ebase + qrow0 + hi * 8 + r) * Dd + h * DHd;
    Og[off + ln] = (_Float16)(O0[r] * inv);
    Og[off + 16 + ln] = (_Float16)(O1[r] * inv);
  }
}

// ---------------- host-side orchestration ----------------
extern "C" void kernel_launch(void* const* d_in, const int* in_sizes, int n_in,
                              void* d_out, int out_size, void* d_ws, size_t ws_size,
                              hipStream_t stream) {
  (void)in_sizes; (void)n_in; (void)out_size; (void)ws_size;
  const float* X = (const float*)d_in[0];
  const int* ei = (const int*)d_in[1];
  const int* srcI = ei;
  const int* tgtI = ei + Bb * Mm;
  // d_in[2] = batch_mapping (structure is known: M edges per graph) -- unused
  const float* Wq = (const float*)d_in[3];  const float* bq = (const float*)d_in[4];
  const float* Wk = (const float*)d_in[5];  const float* bk = (const float*)d_in[6];
  const float* Wv = (const float*)d_in[7];  const float* bv = (const float*)d_in[8];
  const float* Wo = (const float*)d_in[9];  const float* bo = (const float*)d_in[10];
  const float* ln1s = (const float*)d_in[11]; const float* ln1b = (const float*)d_in[12];
  const float* ln2s = (const float*)d_in[13]; const float* ln2b = (const float*)d_in[14];
  const float* W1 = (const float*)d_in[15]; const float* b1 = (const float*)d_in[16];
  const float* W2 = (const float*)d_in[17]; const float* b2 = (const float*)d_in[18];
  const float* W3 = (const float*)d_in[19]; const float* b3 = (const float*)d_in[20];

  char* ws = (char*)d_ws;
  size_t off = 0;
  auto alloc = [&](size_t bytes) -> char* {
    char* p = ws + off;
    off += (bytes + 255) & ~(size_t)255;
    return p;
  };
  _Float16* hWq = (_Float16*)alloc((size_t)Dd * Dd * 2);   // [N][K] transposed
  _Float16* hWk = (_Float16*)alloc((size_t)Dd * Dd * 2);
  _Float16* hWv = (_Float16*)alloc((size_t)Dd * Dd * 2);
  _Float16* hWo = (_Float16*)alloc((size_t)Dd * Dd * 2);
  _Float16* hW1 = (_Float16*)alloc((size_t)Dd * HIDh * 2);
  _Float16* hW2 = (_Float16*)alloc((size_t)HIDh * HIDh * 2);
  _Float16* hW3 = (_Float16*)alloc((size_t)HIDh * Dd * 2);
  _Float16* hXn = (_Float16*)alloc((size_t)NROWS * Dd * 2);
  _Float16* hQ  = (_Float16*)alloc((size_t)NROWS * Dd * 2);
  _Float16* hK  = (_Float16*)alloc((size_t)NROWS * Dd * 2);
  _Float16* hV  = (_Float16*)alloc((size_t)NROWS * Dd * 2);
  _Float16* hO  = (_Float16*)alloc((size_t)NROWS * Dd * 2);
  float*    fOut = (float*)alloc((size_t)NROWS * Dd * 4);
  _Float16* hH  = (_Float16*)alloc((size_t)NROWS * Dd * 2);
  _Float16* hH1 = (_Float16*)alloc((size_t)NROWS * HIDh * 2);
  _Float16* hH2 = (_Float16*)alloc((size_t)NROWS * HIDh * 2);

  auto cvt = [&](const float* s, _Float16* d, int K, int N) {
    cvt_transpose<<<(K * N + 255) / 256, 256, 0, stream>>>(s, d, K, N);
  };
  cvt(Wq, hWq, Dd, Dd);
  cvt(Wk, hWk, Dd, Dd);
  cvt(Wv, hWv, Dd, Dd);
  cvt(Wo, hWo, Dd, Dd);
  cvt(W1, hW1, Dd, HIDh);
  cvt(W2, hW2, HIDh, HIDh);
  cvt(W3, hW3, HIDh, Dd);

  ln_kernel<<<NROWS / 8, 256, 0, stream>>>(X, ln1s, ln1b, hXn);

  // waves = (NROWS/16) * (N/64); blocks = waves/8
  const int gridN256 = (NROWS / 16) * (Dd / 64) / 8;    // 256
  const int gridN512 = (NROWS / 16) * (HIDh / 64) / 8;  // 512
  gemm16<0, Dd><<<gridN256, 256, 0, stream>>>(hXn, hWq, bq, nullptr, hQ, Dd);
  gemm16<0, Dd><<<gridN256, 256, 0, stream>>>(hXn, hWk, bk, nullptr, hK, Dd);
  gemm16<0, Dd><<<gridN256, 256, 0, stream>>>(hXn, hWv, bv, nullptr, hV, Dd);

  attn_kernel<<<Bb * Hh * 8, 128, 0, stream>>>(hQ, hK, hV, srcI, tgtI, hO);

  gemm16<2, Dd><<<gridN256, 256, 0, stream>>>(hO, hWo, bo, X, fOut, Dd);

  ln_kernel<<<NROWS / 8, 256, 0, stream>>>(fOut, ln2s, ln2b, hH);

  gemm16<1, Dd><<<gridN512, 256, 0, stream>>>(hH, hW1, b1, nullptr, hH1, HIDh);
  gemm16<1, HIDh><<<gridN512, 256, 0, stream>>>(hH1, hW2, b2, nullptr, hH2, HIDh);
  gemm16<2, HIDh><<<gridN256, 256, 0, stream>>>(hH2, hW3, b3, fOut, (float*)d_out, Dd);
}
